// TriangleAttentionStartingNode_45088566673706
// MI455X (gfx1250) — compile-verified, matrix operand order checked by reading, not run on hardware
//
#include <hip/hip_runtime.h>
#include <stdint.h>

typedef __attribute__((ext_vector_type(16))) _Float16 v16h;
typedef __attribute__((ext_vector_type(8)))  _Float16 v8h;
typedef __attribute__((ext_vector_type(8)))  float    v8f;
typedef __attribute__((ext_vector_type(4)))  int      v4i;

#define NSEQ 256
#define CDIM 128
#define NHEAD 4
#define CHD 32

#if defined(__has_builtin)
#if __has_builtin(__builtin_amdgcn_global_load_async_to_lds_b128)
#define ASYNC_BUILTIN 1
#endif
#if __has_builtin(__builtin_amdgcn_s_wait_asynccnt)
#define WAIT_ASYNC_BUILTIN 1
#endif
#if __has_builtin(__builtin_amdgcn_update_dpp)
#define HAVE_DPP 1
#endif
#endif

static __device__ __forceinline__ v16h mk16(v8h lo, v8h hi) {
  v16h r;
#pragma unroll
  for (int j = 0; j < 8; ++j) { r[j] = lo[j]; r[j + 8] = hi[j]; }
  return r;
}

static __device__ __forceinline__ v8f wmma_f16(v16h a, v16h b, v8f c) {
  // D = A(16x32 f16) * B(32x16 f16) + C(16x16 f32)
  return __builtin_amdgcn_wmma_f32_16x16x32_f16(false, a, false, b, (short)0, c,
                                                false, false);
}

// Async global->LDS copy of 16 bytes per lane (CDNA5 ASYNCcnt path).
// Builtin signature: (int4 AS1* gsrc, int4 AS3* ldst, imm offset, imm cpol).
static __device__ __forceinline__ void async_copy_b128(const void* g, void* l) {
#if ASYNC_BUILTIN
  __builtin_amdgcn_global_load_async_to_lds_b128(
      (__attribute__((address_space(1))) v4i*)g,
      (__attribute__((address_space(3))) v4i*)l, 0, 0);
#else
  unsigned lo = (unsigned)(uintptr_t)l;       // low 32 bits = LDS byte offset
  unsigned long long ga = (unsigned long long)(uintptr_t)g;
  asm volatile("global_load_async_to_lds_b128 %0, %1, off"
               :: "v"(lo), "v"(ga) : "memory");
#endif
}

static __device__ __forceinline__ void wait_async0() {
#if WAIT_ASYNC_BUILTIN
  __builtin_amdgcn_s_wait_asynccnt(0);
#else
  asm volatile("s_wait_asynccnt 0x0" ::: "memory");
#endif
}

// Butterfly xor within 16-lane DPP rows (masks 1/2/4/8 never cross a row).
template <int MASK>
static __device__ __forceinline__ float row_xor(float x) {
#if HAVE_DPP
  int r = __builtin_amdgcn_update_dpp(0, __builtin_bit_cast(int, x),
                                      0x160 | MASK /*row_xmask*/, 0xf, 0xf,
                                      true);
  return __builtin_bit_cast(float, r);
#else
  return __shfl_xor(x, MASK, 32);
#endif
}

static __device__ __forceinline__ float red_max16(float t) {
  t = fmaxf(t, row_xor<1>(t));
  t = fmaxf(t, row_xor<2>(t));
  t = fmaxf(t, row_xor<4>(t));
  t = fmaxf(t, row_xor<8>(t));
  return t;
}

static __device__ __forceinline__ float red_sum16(float t) {
  t += row_xor<1>(t);
  t += row_xor<2>(t);
  t += row_xor<4>(t);
  t += row_xor<8>(t);
  return t;
}

// ---------------------------------------------------------------------------
// Kernel 1: LayerNorm over last dim (C=128), fp32 in, f16 out. 1 wave per row.
// ---------------------------------------------------------------------------
__global__ void __launch_bounds__(256) ln_kernel(const float* __restrict__ z,
                                                 const float* __restrict__ w,
                                                 const float* __restrict__ b,
                                                 _Float16* __restrict__ out) {
  int row  = blockIdx.x * 8 + (threadIdx.x >> 5);
  int lane = threadIdx.x & 31;
  const float4 xv = ((const float4*)(z + (size_t)row * CDIM))[lane];
  float s = xv.x + xv.y + xv.z + xv.w;
#pragma unroll
  for (int off = 16; off >= 1; off >>= 1) s += __shfl_xor(s, off, 32);
  float mu = s * (1.0f / CDIM);
  float dx = xv.x - mu, dy = xv.y - mu, dz = xv.z - mu, dw = xv.w - mu;
  float vs = dx * dx + dy * dy + dz * dz + dw * dw;
#pragma unroll
  for (int off = 16; off >= 1; off >>= 1) vs += __shfl_xor(vs, off, 32);
  float inv = rsqrtf(vs * (1.0f / CDIM) + 1e-5f);
  const float4 wv = ((const float4*)w)[lane];
  const float4 bv = ((const float4*)b)[lane];
  _Float16* orow = out + (size_t)row * CDIM + lane * 4;
  orow[0] = (_Float16)(dx * inv * wv.x + bv.x);
  orow[1] = (_Float16)(dy * inv * wv.y + bv.y);
  orow[2] = (_Float16)(dz * inv * wv.z + bv.z);
  orow[3] = (_Float16)(dw * inv * wv.w + bv.w);
}

// ---------------------------------------------------------------------------
// Kernel 2/4: X(f16, M x 128) @ W(f32->f16, 128x128) tiled 128x128 per block.
// 8 waves; wave owns 16 rows x 128 cols (8 WMMA accumulators, 4 K-steps).
// MODE 0: store f16.  MODE 1: sigmoid(x+bias) -> f16.  MODE 2: x+bias -> f32.
// ---------------------------------------------------------------------------
template <int MODE>
__global__ void __launch_bounds__(256) gemm128_kernel(
    const _Float16* __restrict__ X, const float* __restrict__ W,
    const float* __restrict__ bias, void* __restrict__ outp) {
  __shared__ __align__(16) _Float16 Wt[CDIM][CDIM + 8];  // transposed weights
  int tid = threadIdx.x;
  for (int idx = tid; idx < CDIM * CDIM; idx += 256) {
    int kk = idx >> 7, nn = idx & 127;
    Wt[nn][kk] = (_Float16)W[idx];
  }
  __syncthreads();

  int wave = tid >> 5, lane = tid & 31;
  int laneRow = lane & 15, half = lane >> 4;
  int rowbase = blockIdx.x * 128 + wave * 16;

  v8f acc[8];
#pragma unroll
  for (int n = 0; n < 8; ++n) acc[n] = v8f{0.f, 0.f, 0.f, 0.f, 0.f, 0.f, 0.f, 0.f};

  const _Float16* xrow = X + (size_t)(rowbase + laneRow) * CDIM;
#pragma unroll
  for (int ks = 0; ks < 4; ++ks) {
    const v8h* ap = (const v8h*)(xrow + ks * 32 + half * 8);
    v16h afrag = mk16(ap[0], ap[2]);  // K = ks*32 + {half*8.., 16+half*8..}
#pragma unroll
    for (int n = 0; n < 8; ++n) {
      const _Float16* wp = &Wt[n * 16 + laneRow][ks * 32 + half * 8];
      v16h bfrag = mk16(*(const v8h*)wp, *(const v8h*)(wp + 16));
      acc[n] = wmma_f16(afrag, bfrag, acc[n]);
    }
  }

#pragma unroll
  for (int n = 0; n < 8; ++n) {
    int col = n * 16 + laneRow;
#pragma unroll
    for (int p = 0; p < 8; ++p) {
      int m = rowbase + half * 8 + p;
      float val = acc[n][p];
      if constexpr (MODE == 1) {
        val += bias[col];
        val = 1.0f / (1.0f + __expf(-val));
      }
      if constexpr (MODE == 2) {
        ((float*)outp)[(size_t)m * CDIM + col] = val + bias[col];
      } else {
        ((_Float16*)outp)[(size_t)m * CDIM + col] = (_Float16)val;
      }
    }
  }
}

// ---------------------------------------------------------------------------
// Kernel 3: attention bias projection  z_ln(65536x128 f16) @ Wb(128x4) -> f32
// ---------------------------------------------------------------------------
__global__ void __launch_bounds__(256) biasproj_kernel(
    const _Float16* __restrict__ zln, const float* __restrict__ Wb,
    float* __restrict__ bout) {
  int idx = blockIdx.x * 256 + threadIdx.x;
  int h = idx & 3, row = idx >> 2;
  const _Float16* xr = zln + (size_t)row * CDIM;
  float s = 0.f;
#pragma unroll
  for (int kk = 0; kk < CDIM; ++kk) s += (float)xr[kk] * Wb[kk * 4 + h];
  bout[idx] = s;
}

// ---------------------------------------------------------------------------
// Kernel 5: flash-style attention per (i, h). 8 waves; wave handles 2 query
// tiles of 16. K staged via CDNA5 async global->LDS; V transposed in LDS;
// online softmax reductions use row_xmask DPP (VALU pipe, no DS pressure).
// ---------------------------------------------------------------------------
__global__ void __launch_bounds__(256) attn_kernel(
    const _Float16* __restrict__ q, const _Float16* __restrict__ k,
    const _Float16* __restrict__ v, const _Float16* __restrict__ g,
    const float* __restrict__ bias, const float* __restrict__ pm,
    _Float16* __restrict__ o) {
  __shared__ __align__(16) _Float16 Klds[NSEQ][CHD];   // key-major, 16 KB
  __shared__ __align__(16) _Float16 Vt[CHD][NSEQ];     // transposed,  16 KB
  __shared__ __align__(16) _Float16 Plds[8][16][32];   // per-wave P,   8 KB
  int i = blockIdx.x, h = blockIdx.y;
  int tid = threadIdx.x, wave = tid >> 5, lane = tid & 31;
  int laneRow = lane & 15, half = lane >> 4;

  {  // stage K (async DMA to LDS) and V (transposed, via VGPRs); 1 key/thread
    int key = tid;
    const _Float16* kr = k + ((size_t)(i * NSEQ + key) * CDIM + h * CHD);
    const _Float16* vr = v + ((size_t)(i * NSEQ + key) * CDIM + h * CHD);
#pragma unroll
    for (int j = 0; j < 4; ++j)
      async_copy_b128(kr + j * 8, &Klds[key][j * 8]);
#pragma unroll
    for (int c = 0; c < CHD; ++c) Vt[c][key] = vr[c];
    wait_async0();
  }
  __syncthreads();

  const float scale = 0.1767766952966369f;  // 1/sqrt(32)
  for (int qt = wave; qt < 16; qt += 8) {
    int q0 = qt * 16;
    const _Float16* qr =
        q + ((size_t)(i * NSEQ + q0 + laneRow) * CDIM + h * CHD);
    v16h qfrag =
        mk16(*(const v8h*)(qr + half * 8), *(const v8h*)(qr + 16 + half * 8));

    float m[8], l[8];
    v8f oacc[2];
#pragma unroll
    for (int p = 0; p < 8; ++p) { m[p] = -1e30f; l[p] = 0.f; }
    oacc[0] = v8f{0.f, 0.f, 0.f, 0.f, 0.f, 0.f, 0.f, 0.f};
    oacc[1] = oacc[0];

    for (int kk = 0; kk < 8; ++kk) {  // 32 keys per chunk
      float sc[2][8];
#pragma unroll
      for (int sub = 0; sub < 2; ++sub) {
        int keybase = kk * 32 + sub * 16;
        int key = keybase + laneRow;
        const _Float16* kb = &Klds[key][half * 8];
        v16h kfrag = mk16(*(const v8h*)kb, *(const v8h*)(kb + 16));
        v8f zero = v8f{0.f, 0.f, 0.f, 0.f, 0.f, 0.f, 0.f, 0.f};
        v8f sf = wmma_f16(qfrag, kfrag, zero);
        float mterm = 1e9f * (pm[i * NSEQ + key] - 1.0f);
#pragma unroll
        for (int p = 0; p < 8; ++p) {
          int qa = q0 + half * 8 + p;
          sc[sub][p] =
              sf[p] * scale + bias[((size_t)qa * NSEQ + key) * 4 + h] + mterm;
        }
      }
      // online softmax update (DPP butterfly within each 16-lane half)
#pragma unroll
      for (int p = 0; p < 8; ++p) {
        float t = red_max16(fmaxf(sc[0][p], sc[1][p]));
        float mn = fmaxf(m[p], t);
        float alpha = __expf(m[p] - mn);
        m[p] = mn;
        float e0 = __expf(sc[0][p] - mn);
        float e1 = __expf(sc[1][p] - mn);
        float rs = red_sum16(e0 + e1);
        l[p] = l[p] * alpha + rs;
        int qrow = half * 8 + p;
        Plds[wave][qrow][laneRow]      = (_Float16)e0;
        Plds[wave][qrow][laneRow + 16] = (_Float16)e1;
        oacc[0][p] *= alpha;
        oacc[1][p] *= alpha;
      }
#if defined(__has_builtin)
#if __has_builtin(__builtin_amdgcn_s_wait_dscnt)
      __builtin_amdgcn_s_wait_dscnt(0);
#endif
#endif
      // O += P(16x32) @ V(32x32) via two N-tiles
      const _Float16* pb = &Plds[wave][laneRow][half * 8];
      v16h pfrag = mk16(*(const v8h*)pb, *(const v8h*)(pb + 16));
#pragma unroll
      for (int n = 0; n < 2; ++n) {
        const _Float16* vb = &Vt[n * 16 + laneRow][kk * 32 + half * 8];
        v16h vfrag = mk16(*(const v8h*)vb, *(const v8h*)(vb + 16));
        oacc[n] = wmma_f16(pfrag, vfrag, oacc[n]);
      }
    }

    // epilogue: 1/l and sigmoid-gate, store f16
#pragma unroll
    for (int n = 0; n < 2; ++n) {
#pragma unroll
      for (int p = 0; p < 8; ++p) {
        int qa = q0 + half * 8 + p;
        int ch = n * 16 + laneRow;
        size_t off = (size_t)(i * NSEQ + qa) * CDIM + h * CHD + ch;
        float val = oacc[n][p] / l[p];
        o[off] = (_Float16)(val * (float)g[off]);
      }
    }
  }
}

// ---------------------------------------------------------------------------
extern "C" void kernel_launch(void* const* d_in, const int* in_sizes, int n_in,
                              void* d_out, int out_size, void* d_ws,
                              size_t ws_size, hipStream_t stream) {
  const float* z    = (const float*)d_in[0];
  const float* pm   = (const float*)d_in[1];
  const float* ln_w = (const float*)d_in[2];
  const float* ln_b = (const float*)d_in[3];
  const float* Wq   = (const float*)d_in[4];
  const float* Wk   = (const float*)d_in[5];
  const float* Wv   = (const float*)d_in[6];
  const float* Wb   = (const float*)d_in[7];
  const float* Wg   = (const float*)d_in[8];
  const float* bg   = (const float*)d_in[9];
  const float* Wo   = (const float*)d_in[10];
  const float* bo   = (const float*)d_in[11];

  const size_t ROWS = (size_t)NSEQ * NSEQ;                       // 65536
  const size_t MATB = ROWS * CDIM * sizeof(_Float16);            // 16 MB
  char* ws = (char*)d_ws;
  _Float16* zln   = (_Float16*)(ws + 0 * MATB);
  _Float16* qbuf  = (_Float16*)(ws + 1 * MATB);
  _Float16* kbuf  = (_Float16*)(ws + 2 * MATB);
  _Float16* vbuf  = (_Float16*)(ws + 3 * MATB);
  _Float16* gbuf  = (_Float16*)(ws + 4 * MATB);
  float*    bproj = (float*)   (ws + 5 * MATB);                  // 1 MB
  _Float16* obuf  = zln;  // safe alias: all z_ln readers run before attn

  ln_kernel<<<ROWS / 8, 256, 0, stream>>>(z, ln_w, ln_b, zln);
  gemm128_kernel<0><<<ROWS / 128, 256, 0, stream>>>(zln, Wq, nullptr, qbuf);
  gemm128_kernel<0><<<ROWS / 128, 256, 0, stream>>>(zln, Wk, nullptr, kbuf);
  gemm128_kernel<0><<<ROWS / 128, 256, 0, stream>>>(zln, Wv, nullptr, vbuf);
  gemm128_kernel<1><<<ROWS / 128, 256, 0, stream>>>(zln, Wg, bg, gbuf);
  biasproj_kernel<<<ROWS * 4 / 256, 256, 0, stream>>>(zln, Wb, bproj);
  attn_kernel<<<dim3(NSEQ, NHEAD), 256, 0, stream>>>(qbuf, kbuf, vbuf, gbuf,
                                                     bproj, pm, obuf);
  gemm128_kernel<2><<<ROWS / 128, 256, 0, stream>>>(obuf, Wo, bo, d_out);
}
